// AttnMemoryUpdater_50251117363836
// MI455X (gfx1250) — matrix operand order repaired and checked by reading, not compiled, not run on hardware
//
#include <hip/hip_runtime.h>

typedef __attribute__((ext_vector_type(16))) _Float16 v16h;
typedef __attribute__((ext_vector_type(8)))  _Float16 v8h;
typedef __attribute__((ext_vector_type(4)))  _Float16 v4h;
typedef __attribute__((ext_vector_type(8)))  float    v8f;

#define NT      16      // nodes per block
#define DM      128
#define DMSG    128
#define DCAT    256
#define MAILN   8
#define ROWS    128     // NT*MAILN
#define THREADS 256

// d_ws layout in halves: [wq 128x128 | wk 128x256 | wv 128x256 | mlp 128x128]
#define WQ_OFF   0
#define WK_OFF   16384
#define WV_OFF   49152
#define WMLP_OFF 81920
#define W_TOTAL  98304

// -------- WMMA helpers (layouts per CDNA5 ISA §7.12.2) --------
__device__ __forceinline__ v8f wmma_f16(v16h a, v16h b, v8f c) {
  return __builtin_amdgcn_wmma_f32_16x16x32_f16(false, a, false, b, (short)0, c,
                                                false, false);
}
// A fragment (16x32 f16) from LDS. lanes0-15: K(kk..kk+15) contiguous
//                                 lanes16-31: K(kk+16..kk+31) contiguous
__device__ __forceinline__ v16h ldsA(const _Float16* base, int ldk, int m, int kk, int sel) {
  const v8h lo = *(const v8h*)(base + m * ldk + kk + sel * 16);
  const v8h hi = *(const v8h*)(base + m * ldk + kk + 8 + sel * 16);
  return __builtin_shufflevector(lo, hi, 0,1,2,3,4,5,6,7,8,9,10,11,12,13,14,15);
}
// B fragment (32x16 f16): B[k][n] = W[n][k]; lane holds 16 contiguous k at row n.
__device__ __forceinline__ v16h gblB(const _Float16* W, int ldk, int n, int kk, int sel) {
  return *(const v16h*)(W + n * ldk + kk + sel * 16);
}

// -------- pre-pass: convert fp32 weights -> f16 in d_ws --------
__global__ void cvt_weights(const float* __restrict__ wq, const float* __restrict__ wk,
                            const float* __restrict__ wv, const float* __restrict__ wm,
                            _Float16* __restrict__ o) {
  int i = blockIdx.x * THREADS + threadIdx.x;
  if (i >= W_TOTAL) return;
  float v;
  if (i < WK_OFF)        v = wq[i];
  else if (i < WV_OFF)   v = wk[i - WK_OFF];
  else if (i < WMLP_OFF) v = wv[i - WV_OFF];
  else                   v = wm[i - WMLP_OFF];
  o[i] = (_Float16)v;
}

// -------- main kernel --------
__global__ void __launch_bounds__(THREADS)
attn_mem_update(const int* __restrict__ nodes, const float* __restrict__ times,
                const float* __restrict__ mem, const float* __restrict__ mail,
                const float* __restrict__ mail_time,
                const float* __restrict__ wq_b, const float* __restrict__ wk_b,
                const float* __restrict__ wv_b, const float* __restrict__ mlp_b,
                const float* __restrict__ ln_g, const float* __restrict__ ln_b,
                const float* __restrict__ time_w, const float* __restrict__ time_b,
                const _Float16* __restrict__ Wh, float* __restrict__ out) {
  extern __shared__ char smem[];
  _Float16* mcat  = (_Float16*)(smem);            // 128x256 f16  = 65536 B
  _Float16* kbuf  = (_Float16*)(smem + 65536);    // 128x128 f16  = 32768 B
  _Float16* vbuf  = (_Float16*)(smem + 98304);    // 128x128 f16  = 32768 B
  float*    memf  = (float*)(smem + 131072);      // 16x128 f32   = 8192 B
  float*    qbuf  = (float*)(smem + 139264);      // 16x128 f32   = 8192 B
  _Float16* ah    = (_Float16*)(smem + 147456);   // 16x128 f16   = 4096 B
  float*    params= (float*)(smem + 151552);      // 8x128 f32    = 4096 B
  float*    dts   = (float*)(smem + 155648);      // 128 f32      = 512 B
  float*    attw  = (float*)(smem + 156160);      // 16x2x8 f32   = 1024 B
  float*    smu   = (float*)(smem + 157184);      // 16 f32
  float*    srs   = (float*)(smem + 157248);      // 16 f32
  int*      nid   = (int*)(smem + 157312);        // 16 i32
  float*    tvalS = (float*)(smem + 157376);      // 16 f32       -> total 157440 B

  const int tid  = threadIdx.x;
  const int lane = tid & 31;
  const int wv   = tid >> 5;       // wave id 0..7
  const int sel  = (lane >> 4) & 1;
  const int ln16 = lane & 15;
  const int s0   = blockIdx.x * NT;

  // ---- P1: node ids / times / params ----
  if (tid < NT) { nid[tid] = nodes[s0 + tid]; tvalS[tid] = times[s0 + tid]; }
  if (tid < 128) {
    params[      tid] = wq_b[tid];  params[128 + tid] = wk_b[tid];
    params[256 + tid] = wv_b[tid];  params[384 + tid] = mlp_b[tid];
    params[512 + tid] = ln_g[tid];  params[640 + tid] = ln_b[tid];
    params[768 + tid] = time_w[tid];params[896 + tid] = time_b[tid];
  }
  __syncthreads();

  // ---- P2: gather mem rows (f32 + f16), dt, mail msg part ----
  for (int i = tid; i < 512; i += THREADS) {           // 16x128 as float4
    int s = i >> 5, q = i & 31;
    const float4 mv = ((const float4*)(mem + (size_t)nid[s] * DM))[q];
    int o = s * DM + q * 4;
    memf[o] = mv.x; memf[o + 1] = mv.y; memf[o + 2] = mv.z; memf[o + 3] = mv.w;
    v4h h = {(_Float16)mv.x, (_Float16)mv.y, (_Float16)mv.z, (_Float16)mv.w};
    *(v4h*)(ah + o) = h;
  }
  for (int i = tid; i < ROWS; i += THREADS) {
    int s = i >> 3, sl = i & 7;
    const size_t mrow = (size_t)nid[s] * MAILN + sl;
    dts[i] = tvalS[s] - mail_time[mrow];
    __builtin_prefetch(mail + mrow * DMSG, 0, 0);      // global_prefetch_b8
  }
  for (int i = tid; i < 4096; i += THREADS) {          // 128x128 msg as float4
    int r = i >> 5, q = i & 31;
    const float4 mv =
        ((const float4*)(mail + ((size_t)nid[r >> 3] * MAILN + (r & 7)) * DMSG))[q];
    v4h h = {(_Float16)mv.x, (_Float16)mv.y, (_Float16)mv.z, (_Float16)mv.w};
    *(v4h*)(mcat + r * DCAT + q * 4) = h;
  }
  __syncthreads();

  // ---- P3: time features (VALU cos, co-executes) + Q GEMM (WMMA) ----
  for (int i = tid; i < 16384; i += THREADS) {
    int r = i >> 7, c = i & 127;
    float ang = fmaf(dts[r], params[768 + c], params[896 + c]);
    mcat[r * DCAT + DMSG + c] = (_Float16)__cosf(ang);
  }
  {
    v8f acc = {};
#pragma unroll
    for (int ks = 0; ks < 4; ++ks) {
      v16h a = ldsA(ah, DM, ln16, ks * 32, sel);
      v16h b = gblB(Wh + WQ_OFF, DM, wv * 16 + ln16, ks * 32, sel);
      acc = wmma_f16(a, b, acc);
    }
    const int n = wv * 16 + ln16;
    const float bias = params[n];
#pragma unroll
    for (int j = 0; j < 8; ++j) qbuf[(j + sel * 8) * DM + n] = acc[j] + bias;
  }
  __syncthreads();

  // ---- P4: K and V GEMMs, M=128 rows, K=256; wave wv owns row tile wv ----
  {
    v16h aF[8];
#pragma unroll
    for (int ks = 0; ks < 8; ++ks) aF[ks] = ldsA(mcat, DCAT, wv * 16 + ln16, ks * 32, sel);
#pragma unroll 1
    for (int nt = 0; nt < 8; ++nt) {
      const int n = nt * 16 + ln16;
      v8f accK = {};
#pragma unroll
      for (int ks = 0; ks < 8; ++ks)
        accK = wmma_f16(aF[ks], gblB(Wh + WK_OFF, DCAT, n, ks * 32, sel), accK);
      const float bk = params[128 + n];
#pragma unroll
      for (int j = 0; j < 8; ++j)
        kbuf[(wv * 16 + j + sel * 8) * DM + n] = (_Float16)(accK[j] + bk);
      v8f accV = {};
#pragma unroll
      for (int ks = 0; ks < 8; ++ks)
        accV = wmma_f16(aF[ks], gblB(Wh + WV_OFF, DCAT, n, ks * 32, sel), accV);
      const float bv = params[256 + n];
#pragma unroll
      for (int j = 0; j < 8; ++j)
        vbuf[(wv * 16 + j + sel * 8) * DM + n] = (_Float16)(accV[j] + bv);
    }
  }
  __syncthreads();

  // ---- P5: attention (one thread per (node, head, slot)) ----
  {
    const int s = tid >> 4, h = (tid >> 3) & 1, mm = tid & 7;
    const float* qrow = qbuf + s * DM + h * 64;
    const _Float16* krow = kbuf + (s * MAILN + mm) * DM + h * 64;
    float sc = 0.f;
#pragma unroll 8
    for (int d = 0; d < 64; ++d) sc = fmaf(qrow[d], (float)krow[d], sc);
    sc = sc >= 0.f ? sc : 0.2f * sc;                    // LeakyReLU(0.2)
    attw[tid] = sc;
    __syncthreads();
    float mx = -3.4e38f;
#pragma unroll
    for (int j = 0; j < 8; ++j) mx = fmaxf(mx, attw[s * 16 + h * 8 + j]);
    float sum = 0.f;
#pragma unroll
    for (int j = 0; j < 8; ++j) sum += __expf(attw[s * 16 + h * 8 + j] - mx);
    const float mye = __expf(sc - mx);
    __syncthreads();
    attw[tid] = mye / sum;                              // softmax over slots
  }
  __syncthreads();
  for (int i = tid; i < 2048; i += THREADS) {           // out = attn@V + residual
    int s = i >> 7, c = i & 127, h = c >> 6;
    float acc = 0.f;
#pragma unroll
    for (int mm = 0; mm < 8; ++mm)
      acc = fmaf(attw[s * 16 + h * 8 + mm], (float)vbuf[(s * MAILN + mm) * DM + c], acc);
    memf[i] = acc + memf[i];
  }
  __syncthreads();

  // ---- P6: LayerNorm ----
  if (tid < NT) {
    float mu = 0.f;
    for (int c = 0; c < DM; ++c) mu += memf[tid * DM + c];
    mu *= (1.f / DM);
    float var = 0.f;
    for (int c = 0; c < DM; ++c) { float d = memf[tid * DM + c] - mu; var = fmaf(d, d, var); }
    var *= (1.f / DM);
    smu[tid] = mu;
    srs[tid] = rsqrtf(var + 1e-5f);
  }
  __syncthreads();
  for (int i = tid; i < 2048; i += THREADS) {
    int s = i >> 7, c = i & 127;
    ah[i] = (_Float16)fmaf((memf[i] - smu[s]) * srs[s], params[512 + c], params[640 + c]);
  }
  __syncthreads();

  // ---- P7: MLP GEMM + ReLU -> global ----
  {
    v8f acc = {};
#pragma unroll
    for (int ks = 0; ks < 4; ++ks) {
      v16h a = ldsA(ah, DM, ln16, ks * 32, sel);
      v16h b = gblB(Wh + WMLP_OFF, DM, wv * 16 + ln16, ks * 32, sel);
      acc = wmma_f16(a, b, acc);
    }
    const int n = wv * 16 + ln16;
    const float bias = params[384 + n];
#pragma unroll
    for (int j = 0; j < 8; ++j) {
      const int m = j + sel * 8;
      const float r = acc[j] + bias;
      out[(size_t)(s0 + m) * DM + n] = r > 0.f ? r : 0.f;
    }
  }
}

extern "C" void kernel_launch(void* const* d_in, const int* in_sizes, int n_in,
                              void* d_out, int out_size, void* d_ws, size_t ws_size,
                              hipStream_t stream) {
  const int*   nodes     = (const int*)d_in[0];
  const float* times     = (const float*)d_in[1];
  const float* mem       = (const float*)d_in[2];
  const float* mail      = (const float*)d_in[3];
  const float* mail_time = (const float*)d_in[4];
  const float* time_w    = (const float*)d_in[5];
  const float* time_b    = (const float*)d_in[6];
  const float* wq_w      = (const float*)d_in[7];
  const float* wq_b      = (const float*)d_in[8];
  const float* wk_w      = (const float*)d_in[9];
  const float* wk_b      = (const float*)d_in[10];
  const float* wv_w      = (const float*)d_in[11];
  const float* wv_b      = (const float*)d_in[12];
  const float* mlp_w     = (const float*)d_in[13];
  const float* mlp_b     = (const float*)d_in[14];
  const float* ln_g      = (const float*)d_in[15];
  const float* ln_b      = (const float*)d_in[16];
  float*       out       = (float*)d_out;
  _Float16*    Wh        = (_Float16*)d_ws;

  const int size = in_sizes[0];                 // 65536
  const int smem_bytes = 157440;
  (void)hipFuncSetAttribute((const void*)attn_mem_update,
                            hipFuncAttributeMaxDynamicSharedMemorySize, smem_bytes);

  cvt_weights<<<(W_TOTAL + THREADS - 1) / THREADS, THREADS, 0, stream>>>(
      wq_w, wk_w, wv_w, mlp_w, Wh);
  attn_mem_update<<<size / NT, THREADS, smem_bytes, stream>>>(
      nodes, times, mem, mail, mail_time, wq_b, wk_b, wv_b, mlp_b, ln_g, ln_b,
      time_w, time_b, Wh, out);
}
